// GNU_Model_56186762166753
// MI455X (gfx1250) — matrix operand order repaired and checked by reading, not compiled, not run on hardware
//
#include <hip/hip_runtime.h>
#include <hip/hip_bf16.h>

#define NUM_STEPS 100
#define HIDDEN    8
#define V_DIM     1000
#define Z_DIM     1000
#define NB        500
#define N_EDGES   2000
#define EPS_LN    1e-5f
#define SLOPE     0.01f

typedef float    v2f  __attribute__((ext_vector_type(2)));
typedef float    v8f  __attribute__((ext_vector_type(8)));
typedef _Float16 v16h __attribute__((ext_vector_type(16)));

#if __has_builtin(__builtin_amdgcn_wmma_f32_16x16x4_f32)
#define HAVE_WMMA_F32X4 1
#else
#define HAVE_WMMA_F32X4 0
#endif

// ---------------------------------------------------------------------------
// Kernel 1: az[s*V_DIM + r] = dot(A[s][r][:], z)  for all steps at once.
// Flattened as a (100000 x 1000) x (1000) matvec, tiled 16 rows per wave,
// driven through v_wmma (z lives in column 0 of the B operand; the other 15
// output columns are zero and discarded). Bandwidth-bound on A (400 MB).
// ---------------------------------------------------------------------------
__global__ void az_wmma_kernel(const float* __restrict__ A,
                               const float* __restrict__ z,
                               float* __restrict__ az, int nTiles) {
  const int lane = threadIdx.x & 31;
  const int tile = blockIdx.x * (blockDim.x >> 5) + (threadIdx.x >> 5);
  if (tile >= nTiles) return;                 // wave-uniform exit (EXEC stays full)
  const int m    = lane & 15;                 // row within 16-row tile
  const int half = lane >> 4;                 // 0: K pair {0,1}, 1: K pair {2,3}
  const size_t row = (size_t)tile * 16 + m;
  const float* __restrict__ Arow = A + row * (size_t)Z_DIM;

  v8f c = {};
#if HAVE_WMMA_F32X4
  for (int k = 0; k < Z_DIM; k += 4) {
    v2f a, b;
    a.x = Arow[k + 2 * half];
    a.y = Arow[k + 2 * half + 1];
    // B (4x16): column n = lane&15; only column 0 carries z.
    float z0 = z[k + 2 * half];
    float z1 = z[k + 2 * half + 1];
    b.x = (m == 0) ? z0 : 0.0f;
    b.y = (m == 0) ? z1 : 0.0f;
    c = __builtin_amdgcn_wmma_f32_16x16x4_f32(false, a, false, b,
                                              (short)0, c, false, false);
  }
#else
  // Fallback: codegen-confirmed f16 WMMA (convert on the fly).
  for (int k = 0; k < Z_DIM; k += 32) {
    v16h a, b;
#pragma unroll
    for (int j = 0; j < 8; ++j) {
      int kbase = k + ((j >> 2) * 16) + (half ? 8 : 0) + 2 * (j & 3);
      a[2 * j]     = (_Float16)Arow[kbase];
      a[2 * j + 1] = (_Float16)Arow[kbase + 1];
    }
#pragma unroll
    for (int i = 0; i < 16; ++i)
      b[i] = (m == 0) ? (_Float16)z[k + (i >> 3) * 16 + half * 8 + (i & 7)] : (_Float16)0.0f;
    c = __builtin_amdgcn_wmma_f32_16x16x32_f16(false, a, false, b,
                                               (short)0, c, false, false);
  }
#endif
  // Column 0 of D lives in lane 0 (M=0..7) and lane 16 (M=8..15).
  if (m == 0) {
    size_t base = (size_t)tile * 16 + half * 8;
#pragma unroll
    for (int i = 0; i < 8; ++i) az[base + i] = c[i];
  }
}

// ---------------------------------------------------------------------------
// GCN symmetric normalization: deg = 1 (self loop) + #incoming; dis = deg^-1/2
// ---------------------------------------------------------------------------
__global__ void norm_kernel(const int* __restrict__ ei, float* __restrict__ dis) {
  __shared__ float deg[NB];
  const int tid = threadIdx.x;
  if (tid < NB) deg[tid] = 1.0f;
  __syncthreads();
  for (int e = tid; e < N_EDGES; e += blockDim.x)
    atomicAdd(&deg[ei[N_EDGES + e]], 1.0f);
  __syncthreads();
  if (tid < NB) dis[tid] = deg[tid] > 0.0f ? rsqrtf(deg[tid]) : 0.0f;
}

__global__ void init_kernel(float* __restrict__ vs) {
  int i = blockIdx.x * blockDim.x + threadIdx.x;
  if (i < V_DIM) vs[i] = (i < NB) ? 0.0f : 1.0f;
}

// ---------------------------------------------------------------------------
// 8-layer MixHop GNN on 500 nodes x 2 channels, fully LDS-resident.
// ---------------------------------------------------------------------------
__global__ void gnn_kernel(const float* __restrict__ vs, float* __restrict__ u,
                           const float* __restrict__ dis, const int* __restrict__ ei,
                           const float* __restrict__ W, const float* __restrict__ bW,
                           const float* __restrict__ lng, const float* __restrict__ lnb,
                           int s) {
  __shared__ float X[NB][2];
  __shared__ float Y[NB][2];
  const int tid = threadIdx.x;
  const float* __restrict__ v = vs + (size_t)s * V_DIM;
  if (tid < NB) { X[tid][0] = v[tid]; X[tid][1] = v[NB + tid]; }
  __syncthreads();

  for (int l = 0; l < HIDDEN; ++l) {
    if (tid < NB) { Y[tid][0] = 0.0f; Y[tid][1] = 0.0f; }
    __syncthreads();
    if (tid < NB) {                      // self loop contribution
      float w = dis[tid] * dis[tid];
      atomicAdd(&Y[tid][0], w * X[tid][0]);
      atomicAdd(&Y[tid][1], w * X[tid][1]);
    }
    for (int e = tid; e < N_EDGES; e += blockDim.x) {
      int srcN = ei[e], dstN = ei[N_EDGES + e];
      float w = dis[srcN] * dis[dstN];
      atomicAdd(&Y[dstN][0], w * X[srcN][0]);
      atomicAdd(&Y[dstN][1], w * X[srcN][1]);
    }
    __syncthreads();
    if (tid < NB) {
      const float* Wl = W   + ((size_t)s * HIDDEN + l) * 8;  // (K=2,2,2)
      const float* bl = bW  + ((size_t)s * HIDDEN + l) * 4;  // (K=2,2)
      const float* g  = lng + ((size_t)s * HIDDEN + l) * 2;
      const float* be = lnb + ((size_t)s * HIDDEN + l) * 2;
      float x0 = X[tid][0], x1 = X[tid][1];
      float y0 = Y[tid][0], y1 = Y[tid][1];
      float o0 = Wl[0] * x0 + Wl[1] * x1 + bl[0] + Wl[4] * y0 + Wl[5] * y1 + bl[2];
      float o1 = Wl[2] * x0 + Wl[3] * x1 + bl[1] + Wl[6] * y0 + Wl[7] * y1 + bl[3];
      float mu = 0.5f * (o0 + o1);
      float d0 = o0 - mu, d1 = o1 - mu;
      float r = rsqrtf(0.5f * (d0 * d0 + d1 * d1) + EPS_LN);
      o0 = d0 * r * g[0] + be[0];
      o1 = d1 * r * g[1] + be[1];
      X[tid][0] = o0 > 0.0f ? o0 : SLOPE * o0;
      X[tid][1] = o1 > 0.0f ? o1 : SLOPE * o1;
    }
    __syncthreads();
  }
  if (tid < NB) { u[tid] = X[tid][0]; u[NB + tid] = X[tid][1]; }
}

// ---------------------------------------------------------------------------
// v_{s+1}[r] = az[s][r] + b[s][r] + dot(Bm[s][r][:], u)
// One wave32 per row; float4 streaming loads + prefetch; shuffle reduction.
// ---------------------------------------------------------------------------
__global__ void matvec_kernel(const float* __restrict__ Bm, const float* __restrict__ bvec,
                              const float* __restrict__ az, const float* __restrict__ u,
                              float* __restrict__ vs, int s) {
  const int lane = threadIdx.x & 31;
  const int row  = blockIdx.x * (blockDim.x >> 5) + (threadIdx.x >> 5);
  if (row >= V_DIM) return;
  const float4* __restrict__ Brow =
      (const float4*)(Bm + (size_t)s * V_DIM * V_DIM + (size_t)row * V_DIM);
  const float4* __restrict__ u4 = (const float4*)u;
  float p = 0.0f;
  for (int k = lane; k < V_DIM / 4; k += 32) {
    __builtin_prefetch(Brow + k + 64, 0, 0);   // global_prefetch_b8
    float4 bv = Brow[k];
    float4 uv = u4[k];
    p += bv.x * uv.x + bv.y * uv.y + bv.z * uv.z + bv.w * uv.w;
  }
#pragma unroll
  for (int off = 16; off > 0; off >>= 1) p += __shfl_down(p, off, 32);
  if (lane == 0) {
    size_t o = (size_t)s * V_DIM + row;
    vs[(size_t)(s + 1) * V_DIM + row] = az[o] + bvec[o] + p;
  }
}

__global__ void copy_last_kernel(float* __restrict__ out, const float* __restrict__ vs) {
  int i = blockIdx.x * blockDim.x + threadIdx.x;
  if (i < V_DIM) out[i] = vs[(size_t)NUM_STEPS * V_DIM + i];
}

extern "C" void kernel_launch(void* const* d_in, const int* in_sizes, int n_in,
                              void* d_out, int out_size, void* d_ws, size_t ws_size,
                              hipStream_t stream) {
  const float* z   = (const float*)d_in[0];
  const int*   ei  = (const int*)d_in[1];
  const float* A   = (const float*)d_in[2];
  const float* Bm  = (const float*)d_in[3];
  const float* b   = (const float*)d_in[4];
  const float* W   = (const float*)d_in[5];
  const float* bW  = (const float*)d_in[6];
  const float* lng = (const float*)d_in[7];
  const float* lnb = (const float*)d_in[8];

  float* out = (float*)d_out;          // [0,1000): v_last ; [1000,102000): vs
  float* vs  = out + V_DIM;
  float* ws  = (float*)d_ws;
  float* az  = ws;                              // 100000 f32
  float* u   = ws + NUM_STEPS * V_DIM;          // 1000 f32
  float* dis = u + V_DIM;                       // 500 f32

  init_kernel<<<4, 256, 0, stream>>>(vs);
  norm_kernel<<<1, 512, 0, stream>>>(ei, dis);

  const int tiles = NUM_STEPS * V_DIM / 16;     // 6250
  az_wmma_kernel<<<(tiles + 7) / 8, 256, 0, stream>>>(A, z, az, tiles);

  for (int s = 0; s < NUM_STEPS; ++s) {
    gnn_kernel<<<1, 512, 0, stream>>>(vs, u, dis, ei, W, bW, lng, lnb, s);
    matvec_kernel<<<V_DIM / 8, 256, 0, stream>>>(Bm, b, az, u, vs, s);
  }
  copy_last_kernel<<<4, 256, 0, stream>>>(out, vs);
}
